// QuantConv2d_12962211299565
// MI455X (gfx1250) — 8-bit matrix operand order repaired and checked by reading, compiled, not run on hardware
//
#include <hip/hip_runtime.h>

typedef int v8i __attribute__((ext_vector_type(8)));

#define CIN       128
#define COUTS     256
#define HW        56
#define PIXPERIMG (56 * 56)
#define NPIX      (32 * 56 * 56)
#define XQ_BYTES  (32 * 56 * 56 * 128)
#define BROW      144
#define OROW      20

__device__ __forceinline__ signed char sgn8(float v) {
    return (v > 0.0f) ? (signed char)1 : ((v < 0.0f) ? (signed char)-1 : (signed char)0);
}

__global__ void __launch_bounds__(256) quant_x_nhwc(const float* __restrict__ x,
                                                    signed char* __restrict__ xq) {
    __shared__ __align__(16) signed char t[32 * BROW];
    const int b      = blockIdx.x;
    const int n      = b / 98;
    const int rrbase = (b % 98) * 32;
    const int pix    = threadIdx.x & 31;
    const int cg     = threadIdx.x >> 5;
    #pragma unroll
    for (int it = 0; it < 16; ++it) {
        int c   = it * 8 + cg;
        float v = x[((size_t)n * CIN + c) * PIXPERIMG + rrbase + pix];
        t[pix * BROW + c] = sgn8(v);
    }
    __syncthreads();
    const int p2  = threadIdx.x >> 3;
    const int off = (threadIdx.x & 7) * 16;
    *(int4*)(xq + ((size_t)(n * PIXPERIMG + rrbase + p2)) * CIN + off) =
        *(const int4*)(t + p2 * BROW + off);
}

__global__ void quant_w(const float* __restrict__ wsrc, signed char* __restrict__ wq) {
    int i = blockIdx.x * 256 + threadIdx.x;
    float v = wsrc[i];
    int kw = i % 3; int t = i / 3;
    int kh = t % 3; t /= 3;
    int ci = t % CIN;
    int o  = t / CIN;
    wq[(o * 9 + kh * 3 + kw) * CIN + ci] = sgn8(v);
}

__global__ void __launch_bounds__(256) bconv_wmma(
    const signed char* __restrict__ xq,
    const signed char* __restrict__ wq,
    const float* __restrict__ bias,
    float* __restrict__ out)
{
    __shared__ __align__(16) signed char ldsB[64 * BROW];
    __shared__ __align__(16) float       ldsO[8 * 64 * OROW];

    const int lane = threadIdx.x & 31;
    const int wave = threadIdx.x >> 5;
    const int mBase    = blockIdx.x * 256;
    const int coutBase = blockIdx.y * 64;

    const int col  = lane & 15;
    const int half = lane >> 4;
    const int koff = half * 16;

    const int waveBase = mBase + wave * 32;
    const int n      = waveBase / PIXPERIMG;
    const int rrbase = waveBase % PIXPERIMG;

    const int rr0 = rrbase + col;
    const int rr1 = rr0 + 16;
    const int oh0 = rr0 / HW, ow0 = rr0 % HW;
    const int oh1 = rr1 / HW, ow1 = rr1 % HW;
    const signed char* ximg = xq + (size_t)n * PIXPERIMG * CIN;

    v8i acc[2][4];
    #pragma unroll
    for (int mt = 0; mt < 2; ++mt)
        #pragma unroll
        for (int nt = 0; nt < 4; ++nt)
            acc[mt][nt] = (v8i){0, 0, 0, 0, 0, 0, 0, 0};

    for (int pos = 0; pos < 9; ++pos) {
        __syncthreads();
        #pragma unroll
        for (int it = 0; it < 2; ++it) {
            int idx = (int)threadIdx.x + it * 256;
            int cl  = idx >> 3;
            int off = (idx & 7) * 16;
            *(int4*)(ldsB + cl * BROW + off) =
                *(const int4*)(wq + ((size_t)(coutBase + cl) * 9 + pos) * CIN + off);
        }
        if (pos < 8) {
            int cl  = (int)threadIdx.x >> 3;
            int off = ((int)threadIdx.x & 7) * 16;
            __builtin_prefetch(wq + ((size_t)(coutBase + cl) * 9 + pos + 1) * CIN + off, 0, 3);
        }
        __syncthreads();

        const int kh = pos / 3, kw = pos % 3;
        const int ih0 = oh0 + kh - 1, iw0 = ow0 + kw - 1;
        const int ih1 = oh1 + kh - 1, iw1 = ow1 + kw - 1;
        const bool v0 = ((unsigned)ih0 < (unsigned)HW) && ((unsigned)iw0 < (unsigned)HW);
        const bool v1 = ((unsigned)ih1 < (unsigned)HW) && ((unsigned)iw1 < (unsigned)HW);
        const signed char* ab0 = ximg + ((size_t)ih0 * HW + iw0) * CIN;
        const signed char* ab1 = ximg + ((size_t)ih1 * HW + iw1) * CIN;

        #pragma unroll
        for (int chunk = 0; chunk < 2; ++chunk) {
            v8i a0 = (v8i){0, 0, 0, 0, 0, 0, 0, 0};
            v8i a1 = (v8i){0, 0, 0, 0, 0, 0, 0, 0};
            if (v0) {
                const signed char* ap = ab0 + chunk * 64 + koff;
                int2 q0 = *(const int2*)(ap);
                int2 q1 = *(const int2*)(ap + 8);
                int2 q2 = *(const int2*)(ap + 32);
                int2 q3 = *(const int2*)(ap + 40);
                a0[0] = q0.x; a0[1] = q0.y; a0[2] = q1.x; a0[3] = q1.y;
                a0[4] = q2.x; a0[5] = q2.y; a0[6] = q3.x; a0[7] = q3.y;
            }
            if (v1) {
                const signed char* ap = ab1 + chunk * 64 + koff;
                int2 q0 = *(const int2*)(ap);
                int2 q1 = *(const int2*)(ap + 8);
                int2 q2 = *(const int2*)(ap + 32);
                int2 q3 = *(const int2*)(ap + 40);
                a1[0] = q0.x; a1[1] = q0.y; a1[2] = q1.x; a1[3] = q1.y;
                a1[4] = q2.x; a1[5] = q2.y; a1[6] = q3.x; a1[7] = q3.y;
            }
            #pragma unroll
            for (int nt = 0; nt < 4; ++nt) {
                const signed char* bp = ldsB + (nt * 16 + col) * BROW + chunk * 64 + half * 16;
                int4 q0 = *(const int4*)(bp);
                int4 q1 = *(const int4*)(bp + 32);
                v8i bfrag;
                bfrag[0] = q0.x; bfrag[1] = q0.y; bfrag[2] = q0.z; bfrag[3] = q0.w;
                bfrag[4] = q1.x; bfrag[5] = q1.y; bfrag[6] = q1.z; bfrag[7] = q1.w;
                acc[0][nt] = __builtin_amdgcn_wmma_i32_16x16x64_iu8(
                    true, a0, true, bfrag, acc[0][nt], false, false);
                acc[1][nt] = __builtin_amdgcn_wmma_i32_16x16x64_iu8(
                    true, a1, true, bfrag, acc[1][nt], false, false);
            }
        }
    }

    float* myO = ldsO + wave * 64 * OROW;
    #pragma unroll
    for (int mt = 0; mt < 2; ++mt) {
        #pragma unroll
        for (int nt = 0; nt < 4; ++nt) {
            #pragma unroll
            for (int r = 0; r < 8; ++r) {
                myO[(nt * 16 + col) * OROW + half * 8 + r] = (float)acc[mt][nt][r];
            }
        }
        #pragma unroll
        for (int rnd = 0; rnd < 2; ++rnd) {
            const int cl   = rnd * 32 + lane;
            const int cout = coutBase + cl;
            const float bv = bias[cout];
            const float bs = (bv > 0.0f) ? 1.0f : ((bv < 0.0f) ? -1.0f : 0.0f);
            float* gdst = out + ((size_t)n * COUTS + cout) * PIXPERIMG + rrbase + mt * 16;
            #pragma unroll
            for (int j = 0; j < 4; ++j) {
                float4 v = *(const float4*)(myO + cl * OROW + j * 4);
                v.x += bs; v.y += bs; v.z += bs; v.w += bs;
                *(float4*)(gdst + j * 4) = v;
            }
        }
    }
}

extern "C" void kernel_launch(void* const* d_in, const int* in_sizes, int n_in,
                              void* d_out, int out_size, void* d_ws, size_t ws_size,
                              hipStream_t stream) {
    const float* x    = (const float*)d_in[0];
    const float* wgt  = (const float*)d_in[1];
    const float* bias = (const float*)d_in[2];
    float* out = (float*)d_out;

    signed char* xq = (signed char*)d_ws;
    signed char* wq = (signed char*)d_ws + (size_t)XQ_BYTES;

    quant_x_nhwc<<<dim3(NPIX / 32), dim3(256), 0, stream>>>(x, xq);
    quant_w<<<dim3((COUTS * CIN * 9) / 256), dim3(256), 0, stream>>>(wgt, wq);
    bconv_wmma<<<dim3(NPIX / 256, COUTS / 64), dim3(256), 0, stream>>>(xq, wq, bias, out);
}
